// RNNMLP_85315230368102
// MI455X (gfx1250) — matrix operand order repaired and checked by reading, not compiled, not run on hardware
//
#include <hip/hip_runtime.h>
#include <hip/hip_bf16.h>

// ---- problem constants (match reference) ----
#define B_TOT 256
#define T_LEN 4096
#define I_DIM 63     // input features
#define H_DIM 50     // hidden
#define FC_DIM 64
#define KPAD 128     // K = [x_t (63, pad->64) | h (50, pad->64)]
#define NPAD 64      // N = H padded to 64
#define ROWS 16      // batch rows per workgroup (one 16-row M tile)
#define NWAVES 4     // one wave per 16-col N tile
#define THREADS (NWAVES * 32)
#define NBLOCKS (B_TOT / ROWS)
#define XELEMS (ROWS * I_DIM)   // 1008 x-elements staged per block per step
#define XFULL  (XELEMS / THREADS)        // 7 full iterations
#define XTAIL  (XELEMS - XFULL * THREADS) // 112 threads in the tail slot
#define PF_DIST 24   // cache prefetch distance (timesteps)

typedef __bf16 bf16_t;
typedef bf16_t v16bf __attribute__((ext_vector_type(16)));
typedef float  v8f   __attribute__((ext_vector_type(8)));

// Branch-free tanh: 1 - 2/(exp2(2*log2(e)*x) + 1).
// exp2 overflow -> +inf -> rcp=0 -> +1; underflow -> 0 -> rcp(1)=1 -> -1.
__device__ __forceinline__ float fast_tanh(float v) {
  float e = __builtin_amdgcn_exp2f(v * 2.8853900817779268f);
  float r = __builtin_amdgcn_rcpf(e + 1.0f);
  return __builtin_fmaf(-2.0f, r, 1.0f);
}

// ---------------------------------------------------------------------------
// Prep: pack Wcat[k][n] = (k<63 ? W_ih[n][k] : (64<=k<114 ? W_hh[n][k-64] : 0))
// as bf16, k-major [128][64]; fused bias b_ih+b_hh (fp32, padded to 64).
// ---------------------------------------------------------------------------
__global__ void rnnmlp_prep_kernel(const float* __restrict__ W_ih,
                                   const float* __restrict__ b_ih,
                                   const float* __restrict__ W_hh,
                                   const float* __restrict__ b_hh,
                                   bf16_t* __restrict__ Wcat,
                                   float* __restrict__ biasv) {
  int tid = threadIdx.x;
  for (int idx = tid; idx < KPAD * NPAD; idx += blockDim.x) {
    int k = idx / NPAD, n = idx % NPAD;
    float v = 0.f;
    if (n < H_DIM) {
      if (k < I_DIM)                      v = W_ih[n * I_DIM + k];
      else if (k >= 64 && k < 64 + H_DIM) v = W_hh[n * H_DIM + (k - 64)];
    }
    Wcat[idx] = (bf16_t)v;
  }
  if (tid < NPAD) biasv[tid] = (tid < H_DIM) ? (b_ih[tid] + b_hh[tid]) : 0.f;
}

// ---------------------------------------------------------------------------
// Fused RNN scan + MLP head + argmax. One workgroup per 16 batch rows.
// Double-buffered operand tile -> one barrier per step; x prefetched 2 steps
// ahead through registers, PF_DIST steps ahead into the WGP-level cache.
// ---------------------------------------------------------------------------
__global__ __launch_bounds__(THREADS)
void rnnmlp_scan_kernel(const float* __restrict__ x,
                        const bf16_t* __restrict__ Wcat_g,
                        const float* __restrict__ biasv_g,
                        const float* __restrict__ W1,
                        const float* __restrict__ b1,
                        const float* __restrict__ W2,
                        const float* __restrict__ b2,
                        float* __restrict__ out) {
  __shared__ bf16_t A0buf[ROWS * KPAD];  // [x_t | h] operand tile, bank 0
  __shared__ bf16_t A1buf[ROWS * KPAD];  // bank 1
  __shared__ bf16_t Wsh[KPAD][NPAD];     // packed weights (init only)

  const int tid   = threadIdx.x;
  const int wave  = tid >> 5;            // N tile handled by this wave
  const int lane  = tid & 31;
  const int lrow  = lane & 15;
  const int half  = lane >> 4;           // 0: lanes 0-15, 1: lanes 16-31
  const int bbase = blockIdx.x * ROWS;
  const bool tail_ok = (tid < XTAIL);

  // ---- init LDS ----
  for (int idx = tid; idx < ROWS * KPAD; idx += THREADS) {
    A0buf[idx] = (bf16_t)0.f;            // h0 = 0, pads = 0
    A1buf[idx] = (bf16_t)0.f;
  }
  for (int idx = tid; idx < KPAD * NPAD; idx += THREADS)
    ((bf16_t*)Wsh)[idx] = Wcat_g[idx];
  __syncthreads();

  // ---- persistent B fragments: 32x16 bf16 tile per K-chunk (8 VGPRs each).
  // Lanes 0-15: col n=lane, K 0..7,16..23; lanes 16-31: col n=lane-16, K 8..15,24..31;
  // VGPR j packs K = 8*half + (j>=4 ? 16 : 0) + 2*(j&3), +1.
  v16bf Bf[4];
  {
    const int col = lrow + wave * 16;
#pragma unroll
    for (int c = 0; c < 4; ++c) {
#pragma unroll
      for (int j = 0; j < 8; ++j) {
        int k = c * 32 + half * 8 + ((j >= 4) ? 16 : 0) + 2 * (j & 3);
        Bf[c][2 * j]     = Wsh[k][col];
        Bf[c][2 * j + 1] = Wsh[k + 1][col];
      }
    }
  }
  const float bias_n = biasv_g[lrow + wave * 16];

  // ---- per-thread x gather offsets (u32 elements; time term added via
  // a uniform base pointer -> saddr+vgpr addressing, no per-step VALU math).
  unsigned voff[8];  // global element offset of (row, feature), t=0
  int      widx[8];  // LDS element offset inside the operand tile
#pragma unroll
  for (int it = 0; it < 8; ++it) {
    int idx = tid + it * THREADS;
    if (idx >= XELEMS) idx = tid;        // clamp tail -> always-valid address
    int br = idx / I_DIM, ii = idx - br * I_DIM;
    voff[it] = (unsigned)((bbase + br) * T_LEN) * I_DIM + ii;
    widx[it] = br * KPAD + ii;
  }

  // A-fragment base offset (16-bit A 16x32 layout) and h writeback offset
  // (C/D layout: VGPR r -> row r + 8*half, col lane&15 within this N tile).
  const int aoff = lrow * KPAD + half * 8;
  const int hoff = (half * 8) * KPAD + 64 + lrow + wave * 16;

  float xA[8], xB[8];

  // ---- stage x_0 into bank 0, preload x_1 into xA ----
  {
#pragma unroll
    for (int it = 0; it < 8; ++it) xA[it] = x[voff[it]];
#pragma unroll
    for (int it = 0; it < XFULL; ++it) A0buf[widx[it]] = (bf16_t)xA[it];
    if (tail_ok) A0buf[widx[XFULL]] = (bf16_t)xA[XFULL];
    const float* xs1 = x + I_DIM;  // t = 1
#pragma unroll
    for (int it = 0; it < 8; ++it) xA[it] = xs1[voff[it]];
  }
  __syncthreads();

  // One RNN step: read AR (bank t&1), write h_{t+1} and x_{t+1} into AW,
  // stage XSTG (= x_{t+1}, loaded the previous step), load x_{t+2} into XLD.
#define RNN_STEP(AR, AW, XSTG, XLD, tcur)                                     \
  do {                                                                        \
    if ((tcur) + 2 < T_LEN) {                                                 \
      const float* xs_ = x + (size_t)((tcur) + 2) * I_DIM;                    \
      _Pragma("unroll") for (int it = 0; it < 8; ++it)                        \
          XLD[it] = xs_[voff[it]];                                            \
    }                                                                         \
    if ((tcur) + PF_DIST < T_LEN) {                                           \
      const float* xp_ = x + (size_t)((tcur) + PF_DIST) * I_DIM;              \
      _Pragma("unroll") for (int it = 0; it < 8; ++it)                        \
          __builtin_prefetch(xp_ + voff[it], 0, 3); /* WGP-scope prefetch */  \
    }                                                                         \
    v16bf Af[4];                                                              \
    _Pragma("unroll") for (int c = 0; c < 4; ++c) {                           \
      _Pragma("unroll") for (int i = 0; i < 8; ++i) {                         \
        int kk = aoff + c * 32 + ((i >= 4) ? 16 : 0) + 2 * (i & 3);           \
        Af[c][2 * i]     = AR[kk];                                            \
        Af[c][2 * i + 1] = AR[kk + 1];                                        \
      }                                                                       \
    }                                                                         \
    v8f acc;                                                                  \
    _Pragma("unroll") for (int r = 0; r < 8; ++r) acc[r] = bias_n;            \
    _Pragma("unroll") for (int c = 0; c < 4; ++c)                             \
        acc = __builtin_amdgcn_wmma_f32_16x16x32_bf16(                        \
            false, Af[c], false, Bf[c], (short)0, acc, false, false);         \
    _Pragma("unroll") for (int r = 0; r < 8; ++r)                             \
        AW[hoff + r * KPAD] = (bf16_t)fast_tanh(acc[r]);                      \
    if ((tcur) + 1 < T_LEN) {                                                 \
      _Pragma("unroll") for (int it = 0; it < XFULL; ++it)                    \
          AW[widx[it]] = (bf16_t)XSTG[it];                                    \
      if (tail_ok) AW[widx[XFULL]] = (bf16_t)XSTG[XFULL];                     \
    }                                                                         \
    __syncthreads();                                                          \
  } while (0)

  for (int t = 0; t < T_LEN; t += 2) {
    RNN_STEP(A0buf, A1buf, xA, xB, t);      // even step: read bank0 -> bank1
    RNN_STEP(A1buf, A0buf, xB, xA, t + 1);  // odd step:  read bank1 -> bank0
  }
#undef RNN_STEP
  // Last step (t=4095, odd) wrote h_T into bank 0.

  // ---- MLP head + argmax (O=2 -> compare logits; ties pick class 0) ----
  if (tid < ROWS) {
    float l0 = b2[0], l1 = b2[1];
    for (int j = 0; j < FC_DIM; ++j) {
      float s = b1[j];
      for (int k = 0; k < H_DIM; ++k)
        s += (float)A0buf[tid * KPAD + 64 + k] * W1[j * H_DIM + k];
      s = fmaxf(s, 0.f);
      l0 += W2[j] * s;            // W2[0][j]
      l1 += W2[FC_DIM + j] * s;   // W2[1][j]
    }
    out[bbase + tid] = (l1 > l0) ? 1.f : 0.f;
  }
}

extern "C" void kernel_launch(void* const* d_in, const int* in_sizes, int n_in,
                              void* d_out, int out_size, void* d_ws, size_t ws_size,
                              hipStream_t stream) {
  const float* x    = (const float*)d_in[0];
  const float* W_ih = (const float*)d_in[1];
  const float* b_ih = (const float*)d_in[2];
  const float* W_hh = (const float*)d_in[3];
  const float* b_hh = (const float*)d_in[4];
  const float* W1   = (const float*)d_in[5];
  const float* b1   = (const float*)d_in[6];
  const float* W2   = (const float*)d_in[7];
  const float* b2   = (const float*)d_in[8];
  float* out = (float*)d_out;

  bf16_t* Wcat  = (bf16_t*)d_ws;                                   // 16 KB
  float*  biasv = (float*)((char*)d_ws + KPAD * NPAD * sizeof(bf16_t));

  rnnmlp_prep_kernel<<<1, 256, 0, stream>>>(W_ih, b_ih, W_hh, b_hh, Wcat, biasv);
  rnnmlp_scan_kernel<<<NBLOCKS, THREADS, 0, stream>>>(
      x, Wcat, biasv, W1, b1, W2, b2, out);
}
